// OneGNN_37177236914919
// MI455X (gfx1250) — compile-verified, matrix operand order checked by reading, not run on hardware
//
#include <hip/hip_runtime.h>
#include <hip/hip_bf16.h>

#define N_NODES  50000
#define N_EDGES  600000
#define DIM      128
#define N_GRAPHS 128
#define OUT_DIM  10

typedef __attribute__((ext_vector_type(16))) __bf16 v16bf;
typedef __attribute__((ext_vector_type(8)))  __bf16 v8bf;
typedef __attribute__((ext_vector_type(8)))  float  v8f;
typedef __attribute__((ext_vector_type(4)))  unsigned int u32x4;
typedef __attribute__((ext_vector_type(8)))  int    i32x8;
typedef __attribute__((ext_vector_type(4)))  int    i32x4;

// ---------------- elementwise f32 -> bf16 ----------------
__global__ __launch_bounds__(256) void k_cvt_bf16(const float* __restrict__ in,
                                                  __bf16* __restrict__ out, int n) {
    int i = blockIdx.x * 256 + threadIdx.x;
    if (i < n) out[i] = (__bf16)in[i];
}

// ---------------- pack 128x128 f32 weight into WMMA B-fragment order -------
// Layout: P[((ct*4 + ks)*32 + lane)*16 + i] = bf16( W[K][n] )
//   n = ct*16 + (lane&15), half = lane>>4,
//   K = ks*32 + 8*half + i + (i>=8 ? 8 : 0)
// so each lane's 16 bf16 fragment is one contiguous 32B chunk.
__global__ __launch_bounds__(256) void k_pack_w(const float* __restrict__ W,
                                                __bf16* __restrict__ P) {
    int idx  = blockIdx.x * 256 + threadIdx.x;   // 0..16383
    int i    = idx & 15;
    int lane = (idx >> 4) & 31;
    int ks   = (idx >> 9) & 3;
    int ct   = idx >> 11;
    int n    = ct * 16 + (lane & 15);
    int half = lane >> 4;
    int K    = ks * 32 + 8 * half + i + ((i >= 8) ? 8 : 0);
    P[idx] = (__bf16)W[K * DIM + n];
}

// ---------------- TDM: DMA one packed 32KB weight array into LDS -----------
// 1-D tensor, data_size=4B, 8192 dwords; D# group0/group1 per CDNA5 ISA §8,
// groups 2/3 zero (tile_dim2/3 unused). Issued by one wave, TENSORcnt-tracked.
__device__ __forceinline__ void tdm_load_w(unsigned lds_off, const __bf16* gptr) {
    unsigned long long ga = (unsigned long long)gptr;
    u32x4 g0;
    g0[0] = 1u;                                      // count=1 (valid user D#)
    g0[1] = lds_off;                                 // lds_addr (bytes)
    g0[2] = (unsigned)ga;                            // global_addr[31:0]
    g0[3] = (unsigned)((ga >> 32) & 0x1FFFFFFu)      // global_addr[56:32]
          | (2u << 30);                              // type=2 ("image")
    i32x8 g1;
    g1[0] = (int)(2u << 16);                         // data_size=2 -> 4 bytes
    g1[1] = (int)(8192u << 16);                      // tensor_dim0[15:0] (=8192)
    g1[2] = (int)(1u << 16);                         // tensor_dim0 hi | tensor_dim1=1
    g1[3] = (int)(8192u << 16);                      // tensor_dim1 hi | tile_dim0=8192
    g1[4] = 1;                                       // tile_dim1=1, tile_dim2=0
    g1[5] = 8192;                                    // tensor_dim0_stride lo
    g1[6] = 0;
    g1[7] = 0;
    i32x4 z = {0, 0, 0, 0};
#if defined(__clang_major__) && (__clang_major__ >= 23)
    i32x8 z8 = {0, 0, 0, 0, 0, 0, 0, 0};
    __builtin_amdgcn_tensor_load_to_lds(g0, g1, z, z, z8, 0);
#else
    __builtin_amdgcn_tensor_load_to_lds(g0, g1, z, z, 0);
#endif
}

// ---------------- dual GEMM: C1 = A@W1, C2 = A@W2 (bf16 in, f32 out) -------
// 4 waves / block, one 16-row strip per wave, all 128 output columns.
// Both packed weight arrays staged in LDS once per block via the TDM;
// WMMA B operands then come from ds_load_b128.
__global__ __launch_bounds__(128) void k_gemm_dual(const __bf16* __restrict__ A,
                                                   const __bf16* __restrict__ P1,
                                                   const __bf16* __restrict__ P2,
                                                   float* __restrict__ C1,
                                                   float* __restrict__ C2,
                                                   int nrows) {
    __shared__ __bf16 lds_w[2][16384];               // 2 x 32KB

    if (threadIdx.x == 0) {
        // LDS byte offset = low 32 bits of the flat address (LDS aperture).
        unsigned base = (unsigned)(size_t)(&lds_w[0][0]);
        tdm_load_w(base,          P1);
        tdm_load_w(base + 32768u, P2);
        __builtin_amdgcn_s_wait_tensorcnt(0);
    }
    // Statically-present (never-executed) store: keeps GlobalOpt from folding
    // the LDS loads of a "never stored" shared array. Grid is 782 blocks.
    if (blockIdx.x == 0x7fffffffu) lds_w[0][threadIdx.x] = (__bf16)0.0f;
    __syncthreads();

    int wave = threadIdx.x >> 5;
    int lane = threadIdx.x & 31;
    int tile = blockIdx.x * 4 + wave;
    int row0 = tile * 16;
    if (row0 >= nrows) return;                 // wave-uniform: EXEC stays all-1s
    int m = lane & 15, half = lane >> 4;
    const __bf16* arow = A + (size_t)(row0 + m) * DIM;
    const __bf16* lw1 = &lds_w[0][0];
    const __bf16* lw2 = &lds_w[1][0];

    const v8f vzero = {0.f, 0.f, 0.f, 0.f, 0.f, 0.f, 0.f, 0.f};
    v8f acc1[8], acc2[8];
    #pragma unroll
    for (int c = 0; c < 8; ++c) { acc1[c] = vzero; acc2[c] = vzero; }

    #pragma unroll
    for (int ks = 0; ks < 4; ++ks) {
        // A fragment (16x32 bf16): two contiguous b128 loads per lane
        union { v16bf v; struct { v8bf lo; v8bf hi; } s; } ua;
        const __bf16* pa = arow + ks * 32 + half * 8;
        ua.s.lo = *(const v8bf*)(pa);
        ua.s.hi = *(const v8bf*)(pa + 16);
        v16bf a = ua.v;
        #pragma unroll
        for (int ct = 0; ct < 8; ++ct) {
            int boff = ((ct * 4 + ks) * 32 + lane) * 16;
            v16bf b1 = *(const v16bf*)(lw1 + boff);   // ds_load_b128 x2
            v16bf b2 = *(const v16bf*)(lw2 + boff);
            acc1[ct] = __builtin_amdgcn_wmma_f32_16x16x32_bf16(
                false, a, false, b1, (short)0, acc1[ct], false, false);
            acc2[ct] = __builtin_amdgcn_wmma_f32_16x16x32_bf16(
                false, a, false, b2, (short)0, acc2[ct], false, false);
        }
    }

    // D layout: lane -> n = lane&15, VGPR r -> row = 8*(lane>>4) + r
    #pragma unroll
    for (int ct = 0; ct < 8; ++ct) {
        int n = ct * 16 + m;
        #pragma unroll
        for (int r = 0; r < 8; ++r) {
            int mm = 8 * half + r;
            size_t o = (size_t)(row0 + mm) * DIM + n;
            C1[o] = acc1[ct][r];
            C2[o] = acc2[ct][r];
        }
    }
}

// ---------------- edge scatter: acc[dst] += t[src]  (one wave per edge) ----
__global__ __launch_bounds__(256) void k_scatter(const float4* __restrict__ t,
                                                 const int* __restrict__ src,
                                                 const int* __restrict__ dst,
                                                 float* __restrict__ acc,
                                                 int nedges) {
    int gid  = blockIdx.x * 256 + threadIdx.x;
    int e    = gid >> 5;
    int lane = gid & 31;
    if (e >= nedges) return;
    int s = src[e], d = dst[e];
    float4 v = t[(size_t)s * 32 + lane];
    float* o = acc + (size_t)d * DIM + lane * 4;
    atomicAdd(o + 0, v.x);
    atomicAdd(o + 1, v.y);
    atomicAdd(o + 2, v.z);
    atomicAdd(o + 3, v.w);
}

// ---------------- ReLU + write f32 and bf16 copies -------------------------
__global__ __launch_bounds__(256) void k_relu_cvt(const float* __restrict__ acc,
                                                  float* __restrict__ hf,
                                                  __bf16* __restrict__ hb, int n) {
    int i = blockIdx.x * 256 + threadIdx.x;
    if (i >= n) return;
    float v = acc[i];
    v = v > 0.f ? v : 0.f;
    hf[i] = v;
    hb[i] = (__bf16)v;
}

// ---------------- zero buffer ----------------------------------------------
__global__ __launch_bounds__(256) void k_zero(float* __restrict__ p, int n) {
    int i = blockIdx.x * 256 + threadIdx.x;
    if (i < n) p[i] = 0.f;
}

// ---------------- global add pool: emb[batch[n]] += h[n] -------------------
__global__ __launch_bounds__(256) void k_pool(const float4* __restrict__ hf,
                                              const int* __restrict__ batch,
                                              float* __restrict__ emb, int nnodes) {
    int gid  = blockIdx.x * 256 + threadIdx.x;
    int node = gid >> 5;
    int lane = gid & 31;
    if (node >= nnodes) return;
    int g = batch[node];
    float4 v = hf[(size_t)node * 32 + lane];
    float* o = emb + (size_t)g * DIM + lane * 4;
    atomicAdd(o + 0, v.x);
    atomicAdd(o + 1, v.y);
    atomicAdd(o + 2, v.z);
    atomicAdd(o + 3, v.w);
}

// ---------------- classifier head (tiny, exact fp32) -----------------------
__global__ __launch_bounds__(256) void k_cls1(const float* __restrict__ emb,
                                              const float* __restrict__ Wc1,
                                              const float* __restrict__ bc1,
                                              float* __restrict__ hid) {
    int idx = blockIdx.x * 256 + threadIdx.x;   // 128*128
    int g = idx >> 7, j = idx & 127;
    float s = bc1[j];
    for (int k = 0; k < DIM; ++k) s += emb[g * DIM + k] * Wc1[k * DIM + j];
    hid[idx] = s > 0.f ? s : 0.f;
}

__global__ __launch_bounds__(256) void k_cls2(const float* __restrict__ hid,
                                              const float* __restrict__ Wc2,
                                              const float* __restrict__ bc2,
                                              float* __restrict__ out) {
    int idx = blockIdx.x * 256 + threadIdx.x;   // 128*10
    if (idx >= N_GRAPHS * OUT_DIM) return;
    int g = idx / OUT_DIM, o = idx % OUT_DIM;
    float s = bc2[o];
    for (int j = 0; j < DIM; ++j) s += hid[g * DIM + j] * Wc2[j * OUT_DIM + o];
    out[idx] = s;
}

extern "C" void kernel_launch(void* const* d_in, const int* in_sizes, int n_in,
                              void* d_out, int out_size, void* d_ws, size_t ws_size,
                              hipStream_t stream) {
    (void)in_sizes; (void)n_in; (void)out_size; (void)ws_size;

    const float* x     = (const float*)d_in[0];
    const int*   ei    = (const int*)d_in[1];
    const int*   src   = ei;
    const int*   dst   = ei + N_EDGES;
    const int*   batch = (const int*)d_in[2];
    const float* W1[3] = {(const float*)d_in[3], (const float*)d_in[5], (const float*)d_in[7]};
    const float* W2[3] = {(const float*)d_in[4], (const float*)d_in[6], (const float*)d_in[8]};
    const float* Wc1   = (const float*)d_in[9];
    const float* bc1   = (const float*)d_in[10];
    const float* Wc2   = (const float*)d_in[11];
    const float* bc2   = (const float*)d_in[12];
    float*       out   = (float*)d_out;

    // Carve workspace
    char*  ws  = (char*)d_ws;
    size_t off = 0;
    auto alloc = [&](size_t bytes) -> void* {
        void* p = ws + off;
        off = (off + bytes + 255) & ~(size_t)255;
        return p;
    };
    __bf16* hb  = (__bf16*)alloc((size_t)N_NODES * DIM * sizeof(__bf16));
    float*  acc = (float*) alloc((size_t)N_NODES * DIM * sizeof(float));
    float*  tb  = (float*) alloc((size_t)N_NODES * DIM * sizeof(float));
    float*  hf  = (float*) alloc((size_t)N_NODES * DIM * sizeof(float));
    __bf16* Wp[6];
    for (int i = 0; i < 6; ++i) Wp[i] = (__bf16*)alloc((size_t)16384 * sizeof(__bf16));
    float*  emb = (float*)alloc((size_t)N_GRAPHS * DIM * sizeof(float));
    float*  hid = (float*)alloc((size_t)N_GRAPHS * DIM * sizeof(float));

    // Prep: input features -> bf16, weights -> WMMA B-fragment packing
    k_cvt_bf16<<<(N_NODES * DIM + 255) / 256, 256, 0, stream>>>(x, hb, N_NODES * DIM);
    for (int l = 0; l < 3; ++l) {
        k_pack_w<<<64, 256, 0, stream>>>(W1[l], Wp[2 * l]);
        k_pack_w<<<64, 256, 0, stream>>>(W2[l], Wp[2 * l + 1]);
    }

    const int ntiles  = (N_NODES + 15) / 16;   // 3125
    const int gblocks = (ntiles + 3) / 4;      // 782

    for (int l = 0; l < 3; ++l) {
        // acc = h@W1 (self part, full overwrite), tb = h@W2 (messages)
        k_gemm_dual<<<gblocks, 128, 0, stream>>>(hb, Wp[2 * l], Wp[2 * l + 1],
                                                 acc, tb, N_NODES);
        // acc[dst] += tb[src]
        k_scatter<<<((size_t)N_EDGES * 32 + 255) / 256, 256, 0, stream>>>(
            (const float4*)tb, src, dst, acc, N_EDGES);
        // h = relu(acc) in both f32 and bf16
        k_relu_cvt<<<(N_NODES * DIM + 255) / 256, 256, 0, stream>>>(
            acc, hf, hb, N_NODES * DIM);
    }

    // Pooling + classifier
    k_zero<<<(N_GRAPHS * DIM + 255) / 256, 256, 0, stream>>>(emb, N_GRAPHS * DIM);
    k_pool<<<((size_t)N_NODES * 32 + 255) / 256, 256, 0, stream>>>(
        (const float4*)hf, batch, emb, N_NODES);
    k_cls1<<<(N_GRAPHS * DIM + 255) / 256, 256, 0, stream>>>(emb, Wc1, bc1, hid);
    k_cls2<<<(N_GRAPHS * OUT_DIM + 255) / 256, 256, 0, stream>>>(hid, Wc2, bc2, out);
}
